// NewAttention_64682207478094
// MI455X (gfx1250) — compile-verified, matrix operand order checked by reading, not run on hardware
//
#include <hip/hip_runtime.h>

// ---------------------------------------------------------------------------
// Problem constants (from reference): B=8, M=4096, K=24, DIM=128, H=8
// ---------------------------------------------------------------------------
#define NB    8
#define NM    4096
#define NK    24
#define NDIM  128
#define NH    8
#define NROWS 786432L   // B*M*K
#define NPTS  32768L    // B*M
#define BN_EPS 1e-5f

typedef __attribute__((ext_vector_type(16))) __bf16 v16bf;
typedef __attribute__((ext_vector_type(8)))  float  v8f;

union AFrag { v16bf v; unsigned short u[16]; };
union CFrag { v8f   v; float f[8]; };

__device__ __forceinline__ unsigned short f2bf_bits(float f) {
  union { float f; unsigned u; } c; c.f = f;
  unsigned u = c.u + 0x7FFFu + ((c.u >> 16) & 1u);   // round-to-nearest-even
  return (unsigned short)(u >> 16);
}

// Channel index inside a 32-wide K-chunk for WMMA 16-bit A/B fragments
// (ISA 7.12.2): lane<16 -> base 0, lane>=16 -> base 8; elems 0..7 = K base+i,
// elems 8..15 = K base+16+(i-8).
__device__ __forceinline__ int frag_ch(int base, int i) {
  return base + i + (i >= 8 ? 8 : 0);
}

// ---------------------------------------------------------------------------
// Pack a row-major f32 weight [K x Nreal] into bf16 WMMA B-fragments:
// frag[((kt*NT + nt)*32 + lane)*16 + e], NT = Npad/16.  Zero-pads n >= Nreal.
// ---------------------------------------------------------------------------
__global__ void conv_wfrag_kernel(const float* __restrict__ W,
                                  unsigned short* __restrict__ frag,
                                  int Kdim, int Npad, int Nreal, int nelems) {
  int idx = blockIdx.x * blockDim.x + threadIdx.x;
  if (idx >= nelems) return;
  int e    = idx & 15;
  int lane = (idx >> 4) & 31;
  int f    = idx >> 9;           // fragment id = kt*NT + nt
  int NT   = Npad >> 4;
  int nt   = f % NT;
  int kt   = f / NT;
  int n    = nt * 16 + (lane & 15);
  int base = (lane < 16) ? 0 : 8;
  int k    = kt * 32 + frag_ch(base, e);
  float w  = (n < Nreal && k < Kdim) ? W[k * Nreal + n] : 0.f;
  frag[idx] = f2bf_bits(w);
}

// ---------------------------------------------------------------------------
// Per-channel sum/sumsq of concat(key,pos) over all rows (256 channels).
// Thread c owns channel c; block-strided over rows; coalesced across lanes.
// ---------------------------------------------------------------------------
__global__ void stats_input_kernel(const float* __restrict__ key,
                                   const float* __restrict__ pos,
                                   float* __restrict__ sum1,
                                   float* __restrict__ ss1, long nrows) {
  int c = threadIdx.x;  // 0..255
  const float* src = (c < NDIM) ? (key + c) : (pos + (c - NDIM));
  float s = 0.f, q = 0.f;
  for (long r = blockIdx.x; r < nrows; r += gridDim.x) {
    float x = src[r * NDIM];
    s += x; q += x * x;
  }
  atomicAdd(&sum1[c], s);
  atomicAdd(&ss1[c], q);
}

// ---------------------------------------------------------------------------
// (sum,sumsq) -> per-channel BN scale a = g*rsqrt(var+eps), shift c = b - m*a
// ---------------------------------------------------------------------------
__global__ void finalize_stats_kernel(const float* __restrict__ sum,
                                      const float* __restrict__ ss,
                                      const float* __restrict__ g,
                                      const float* __restrict__ b,
                                      float* __restrict__ a,
                                      float* __restrict__ c,
                                      int nch, float invn) {
  int i = threadIdx.x;
  if (i < nch) {
    float mean = sum[i] * invn;
    float var  = ss[i] * invn - mean * mean;
    float s    = g[i] * rsqrtf(var + BN_EPS);
    a[i] = s;
    c[i] = b[i] - mean * s;
  }
}

// ---------------------------------------------------------------------------
// GEMM1: h1 = relu(bn1(concat(key,pos))) @ W1 + bias1   [786432 x 256]@[256 x 128]
// One wave per 16-row tile; 8 K-chunks x 8 N-tiles = 64 bf16 WMMAs per wave.
// Also accumulates per-channel sum/sumsq of h1 (LDS partial -> global atomic).
// ---------------------------------------------------------------------------
__global__ void gemm1_kernel(const float* __restrict__ key,
                             const float* __restrict__ pos,
                             const float* __restrict__ a1g,
                             const float* __restrict__ c1g,
                             const float* __restrict__ bias1,
                             const unsigned short* __restrict__ w1f,
                             float* __restrict__ h1,
                             float* __restrict__ sum2,
                             float* __restrict__ ss2, int ntiles) {
  __shared__ float s_a[256], s_c[256], s_sum[128], s_ss[128];
  int tid = threadIdx.x;
  if (tid < 128) { s_sum[tid] = 0.f; s_ss[tid] = 0.f; }
  s_a[tid] = a1g[tid]; s_c[tid] = c1g[tid];   // blockDim.x == 256
  __syncthreads();

  int wave = tid >> 5, lane = tid & 31;
  int tile = blockIdx.x * 8 + wave;           // grid is exact: always < ntiles
  int m = lane & 15, base = (lane < 16) ? 0 : 8;
  long row = (long)tile * 16 + m;

  CFrag acc[8];
  for (int nt = 0; nt < 8; ++nt)
    for (int j = 0; j < 8; ++j) acc[nt].f[j] = 0.f;

  for (int kt = 0; kt < 8; ++kt) {
    AFrag av;
    int c0 = kt * 32;
#pragma unroll
    for (int i = 0; i < 16; ++i) {
      int ch = c0 + frag_ch(base, i);
      float x = (ch < NDIM) ? key[row * NDIM + ch] : pos[row * NDIM + ch - NDIM];
      x = x * s_a[ch] + s_c[ch];
      x = x > 0.f ? x : 0.f;
      av.u[i] = f2bf_bits(x);
    }
#pragma unroll
    for (int nt = 0; nt < 8; ++nt) {
      v16bf bv = *(const v16bf*)(w1f + (((kt << 3) + nt) * 32 + lane) * 16);
      acc[nt].v = __builtin_amdgcn_wmma_f32_16x16x32_bf16(
          false, av.v, false, bv, (short)0, acc[nt].v, false, false);
    }
  }

  // D layout: VGPR j holds row j (lanes 0-15) / j+8 (lanes 16-31), N = lane&15
  int nlo = lane & 15, mh = (lane >> 4) * 8;
  long row0 = (long)tile * 16;
  for (int nt = 0; nt < 8; ++nt) {
    int ch = nt * 16 + nlo;
    float b = bias1[ch];
    float s = 0.f, q = 0.f;
#pragma unroll
    for (int j = 0; j < 8; ++j) {
      float v = acc[nt].f[j] + b;
      h1[(row0 + mh + j) * NDIM + ch] = v;
      s += v; q += v * v;
    }
    atomicAdd(&s_sum[ch], s);
    atomicAdd(&s_ss[ch], q);
  }
  __syncthreads();
  if (tid < 128) { atomicAdd(&sum2[tid], s_sum[tid]); atomicAdd(&ss2[tid], s_ss[tid]); }
}

// ---------------------------------------------------------------------------
// Attention: per point p (one wave each):
//   logits[24x8] = relu(bn2(h1[p])) @ W2 + bias2   (2 M-tiles x 4 K-chunks WMMA)
//   softmax over k=24 per head (via LDS), write attn output,
//   x[128] = sum_k attn[k,h]*value[p,k,h*16+d], write x + stats3.
// ---------------------------------------------------------------------------
__global__ void attn_kernel(const float* __restrict__ h1,
                            const float* __restrict__ value,
                            const float* __restrict__ a2g,
                            const float* __restrict__ c2g,
                            const float* __restrict__ bias2,
                            const unsigned short* __restrict__ w2f,
                            float* __restrict__ xout,
                            float* __restrict__ attn_out,
                            float* __restrict__ sum3,
                            float* __restrict__ ss3, int npoints) {
  __shared__ float s_a[128], s_c[128], s_sum[128], s_ss[128];
  __shared__ float s_log[8][32][NH];
  __shared__ float s_attn[8][NK][NH];
  int tid = threadIdx.x;
  if (tid < 128) { s_a[tid] = a2g[tid]; s_c[tid] = c2g[tid]; s_sum[tid] = 0.f; s_ss[tid] = 0.f; }
  __syncthreads();

  int wave = tid >> 5, lane = tid & 31;
  int p = blockIdx.x * 8 + wave;              // grid exact: p < npoints
  int m = lane & 15, base = (lane < 16) ? 0 : 8;
  long rbase = (long)p * NK;

  CFrag acc0, acc1;
  for (int j = 0; j < 8; ++j) { acc0.f[j] = 0.f; acc1.f[j] = 0.f; }

  for (int kt = 0; kt < 4; ++kt) {
    AFrag av0, av1;
    int c0 = kt * 32;
#pragma unroll
    for (int i = 0; i < 16; ++i) {
      int ch = c0 + frag_ch(base, i);
      float sa = s_a[ch], sc = s_c[ch];
      float x0 = h1[(rbase + m) * NDIM + ch] * sa + sc;
      av0.u[i] = f2bf_bits(x0 > 0.f ? x0 : 0.f);
      float x1 = 0.f;                                  // rows 24..31 padded zero
      if (m < NK - 16) {
        x1 = h1[(rbase + 16 + m) * NDIM + ch] * sa + sc;
        x1 = x1 > 0.f ? x1 : 0.f;
      }
      av1.u[i] = f2bf_bits(x1);
    }
    v16bf bv = *(const v16bf*)(w2f + (kt * 32 + lane) * 16);
    acc0.v = __builtin_amdgcn_wmma_f32_16x16x32_bf16(
        false, av0.v, false, bv, (short)0, acc0.v, false, false);
    acc1.v = __builtin_amdgcn_wmma_f32_16x16x32_bf16(
        false, av1.v, false, bv, (short)0, acc1.v, false, false);
  }

  int nlo = lane & 15, mh = (lane >> 4) * 8;
  if (nlo < NH) {
    float b = bias2[nlo];
    for (int j = 0; j < 8; ++j) {
      s_log[wave][mh + j][nlo]      = acc0.f[j] + b;
      s_log[wave][16 + mh + j][nlo] = acc1.f[j] + b;  // rows 24..31 ignored
    }
  }
  __syncthreads();

  int b_idx = p >> 12;          // / 4096
  int mm    = p & 4095;
  if (lane < NH) {
    int h = lane;
    float mx = -3.4e38f;
    for (int k = 0; k < NK; ++k) { float l = s_log[wave][k][h]; mx = l > mx ? l : mx; }
    float sum = 0.f;
    for (int k = 0; k < NK; ++k) {
      float e = __expf(s_log[wave][k][h] - mx);
      s_attn[wave][k][h] = e; sum += e;
    }
    float inv = 1.f / sum;
    long ob = (((long)(b_idx * NH + h)) * NM + mm) * NK;
    for (int k = 0; k < NK; ++k) {
      float a = s_attn[wave][k][h] * inv;
      s_attn[wave][k][h] = a;
      attn_out[ob + k] = a;
    }
  }
  __syncthreads();

#pragma unroll
  for (int q = 0; q < 4; ++q) {
    int ch = lane + 32 * q;
    int h = ch >> 4;
    float s = 0.f;
    for (int k = 0; k < NK; ++k)
      s += s_attn[wave][k][h] * value[(rbase + k) * NDIM + ch];
    xout[(long)p * NDIM + ch] = s;
    atomicAdd(&s_sum[ch], s);
    atomicAdd(&s_ss[ch], s * s);
  }
  __syncthreads();
  if (tid < 128) { atomicAdd(&sum3[tid], s_sum[tid]); atomicAdd(&ss3[tid], s_ss[tid]); }
}

// ---------------------------------------------------------------------------
// GEMM3: out = relu(bn3(x)) @ W3 + bias3    [32768 x 128]@[128 x 128]
// ---------------------------------------------------------------------------
__global__ void gemm3_kernel(const float* __restrict__ xin,
                             const float* __restrict__ a3g,
                             const float* __restrict__ c3g,
                             const float* __restrict__ bias3,
                             const unsigned short* __restrict__ w3f,
                             float* __restrict__ out, int ntiles) {
  __shared__ float s_a[128], s_c[128];
  int tid = threadIdx.x;
  if (tid < 128) { s_a[tid] = a3g[tid]; s_c[tid] = c3g[tid]; }
  __syncthreads();

  int wave = tid >> 5, lane = tid & 31;
  int tile = blockIdx.x * 8 + wave;           // grid exact
  int m = lane & 15, base = (lane < 16) ? 0 : 8;
  long row = (long)tile * 16 + m;

  CFrag acc[8];
  for (int nt = 0; nt < 8; ++nt)
    for (int j = 0; j < 8; ++j) acc[nt].f[j] = 0.f;

  for (int kt = 0; kt < 4; ++kt) {
    AFrag av;
    int c0 = kt * 32;
#pragma unroll
    for (int i = 0; i < 16; ++i) {
      int ch = c0 + frag_ch(base, i);
      float x = xin[row * NDIM + ch] * s_a[ch] + s_c[ch];
      av.u[i] = f2bf_bits(x > 0.f ? x : 0.f);
    }
#pragma unroll
    for (int nt = 0; nt < 8; ++nt) {
      v16bf bv = *(const v16bf*)(w3f + (((kt << 3) + nt) * 32 + lane) * 16);
      acc[nt].v = __builtin_amdgcn_wmma_f32_16x16x32_bf16(
          false, av.v, false, bv, (short)0, acc[nt].v, false, false);
    }
  }

  int nlo = lane & 15, mh = (lane >> 4) * 8;
  long row0 = (long)tile * 16;
  for (int nt = 0; nt < 8; ++nt) {
    int ch = nt * 16 + nlo;
    float b = bias3[ch];
#pragma unroll
    for (int j = 0; j < 8; ++j)
      out[(row0 + mh + j) * NDIM + ch] = acc[nt].f[j] + b;
  }
}

// ---------------------------------------------------------------------------
// Host-side orchestration (graph-capture safe: memsetAsync + kernels only).
// ---------------------------------------------------------------------------
extern "C" void kernel_launch(void* const* d_in, const int* in_sizes, int n_in,
                              void* d_out, int out_size, void* d_ws, size_t ws_size,
                              hipStream_t stream) {
  // inputs (setup_inputs order); query (d_in[0]) is unused by the reference
  const float* key   = (const float*)d_in[1];
  const float* pos   = (const float*)d_in[2];
  const float* value = (const float*)d_in[3];
  const float* g1    = (const float*)d_in[4];
  const float* b1    = (const float*)d_in[5];
  const float* W1    = (const float*)d_in[6];
  const float* bias1 = (const float*)d_in[7];
  const float* g2    = (const float*)d_in[8];
  const float* b2    = (const float*)d_in[9];
  const float* W2    = (const float*)d_in[10];
  const float* bias2 = (const float*)d_in[11];
  const float* g3    = (const float*)d_in[12];
  const float* b3    = (const float*)d_in[13];
  const float* W3    = (const float*)d_in[14];
  const float* bias3 = (const float*)d_in[15];

  float* ws = (float*)d_ws;
  // stats + scale/shift region (zeroed each call)
  float* sum1 = ws;         float* ss1 = ws + 256;
  float* sum2 = ws + 512;   float* ss2 = ws + 640;
  float* sum3 = ws + 768;   float* ss3 = ws + 896;
  float* a1 = ws + 1024;    float* c1 = ws + 1280;
  float* a2 = ws + 1536;    float* c2 = ws + 1664;
  float* a3 = ws + 1792;    float* c3 = ws + 1920;
  // bf16 weight fragments (ushort), 32B-aligned offsets
  unsigned short* w1f = (unsigned short*)(ws + 2048);                  // 32768 ush
  unsigned short* w2f = (unsigned short*)(ws + 2048 + 16384);          //  2048 ush
  unsigned short* w3f = (unsigned short*)(ws + 2048 + 16384 + 1024);   // 16384 ush
  // big buffers
  float* h1   = ws + 27648;                 // 786432*128 f32 (~403 MB)
  float* xbuf = h1 + NROWS * NDIM;          // 32768*128 f32 (16 MB)

  float* out_x    = (float*)d_out;                       // [B,M,128]
  float* out_attn = out_x + (long)NB * NM * NDIM;        // [B,H,M,K,1]

  hipMemsetAsync(ws, 0, 1024 * sizeof(float), stream);

  conv_wfrag_kernel<<<(32768 + 255) / 256, 256, 0, stream>>>(W1, w1f, 256, 128, 128, 32768);
  conv_wfrag_kernel<<<(2048  + 255) / 256, 256, 0, stream>>>(W2, w2f, 128,  16,   8,  2048);
  conv_wfrag_kernel<<<(16384 + 255) / 256, 256, 0, stream>>>(W3, w3f, 128, 128, 128, 16384);

  stats_input_kernel<<<2048, 256, 0, stream>>>(key, pos, sum1, ss1, NROWS);
  finalize_stats_kernel<<<1, 256, 0, stream>>>(sum1, ss1, g1, b1, a1, c1, 256, 1.f / (float)NROWS);

  // 786432 rows / 16 per wave = 49152 tiles; 8 waves/block -> 6144 blocks
  gemm1_kernel<<<6144, 256, 0, stream>>>(key, pos, a1, c1, bias1, w1f, h1, sum2, ss2, 49152);
  finalize_stats_kernel<<<1, 128, 0, stream>>>(sum2, ss2, g2, b2, a2, c2, 128, 1.f / (float)NROWS);

  // 32768 points, 1 wave each, 8 waves/block -> 4096 blocks
  attn_kernel<<<4096, 256, 0, stream>>>(h1, value, a2, c2, bias2, w2f, xbuf, out_attn, sum3, ss3, (int)NPTS);
  finalize_stats_kernel<<<1, 128, 0, stream>>>(sum3, ss3, g3, b3, a3, c3, 128, 1.f / (float)NPTS);

  // 32768 rows / 16 = 2048 tiles -> 256 blocks
  gemm3_kernel<<<256, 256, 0, stream>>>(xbuf, a3, c3, bias3, w3f, out_x, 2048);
}
